// SelfAttention_12094627906024
// MI455X (gfx1250) — matrix operand order repaired and checked by reading, not compiled, not run on hardware
//
#include <hip/hip_runtime.h>
#include <hip/hip_bf16.h>

// Problem constants (match reference)
#define DIMX   128
#define HEADS  8
#define DH     64
#define INNER  512          // HEADS*DH
#define RROWS  64
#define NRES   512
#define DPAIR  128
#define MROWS  (RROWS*NRES) // 32768 rows of x

typedef __attribute__((ext_vector_type(16))) __bf16 v16bf;
typedef __attribute__((ext_vector_type(8)))  float  v8f;

static __device__ __forceinline__ v8f wmma_bf16(v16bf a, v16bf b, v8f c) {
  // D = A(16x32 bf16) * B(32x16 bf16) + C(16x16 f32)
  return __builtin_amdgcn_wmma_f32_16x16x32_bf16(
      /*neg_a=*/false, a, /*neg_b=*/false, b,
      /*c_mod=*/(short)0, c, /*reuse_a=*/false, /*reuse_b=*/false);
}

// ---- Fragment loaders -------------------------------------------------------
// A 16x32 bf16: lane m=lane&15 holds row m; elems [0..15]=K[base..base+16),
// base=(lane>>4)*16 (contiguous, matches B loaders).
static __device__ __forceinline__ v16bf load_fragA(const __bf16* A, int lda,
                                                   int m0, int k0) {
  const int lane = threadIdx.x & 31;
  const int m = lane & 15, sel = lane >> 4;
  const __bf16* p = A + (size_t)(m0 + m) * lda + k0 + sel * 16;
  union { uint4 u; __bf16 h[8]; } u0, u1;
  u0.u = *reinterpret_cast<const uint4*>(p);
  u1.u = *reinterpret_cast<const uint4*>(p + 8);
  v16bf f;
#pragma unroll
  for (int e = 0; e < 8; ++e) { f[e] = u0.h[e]; f[8 + e] = u1.h[e]; }
  return f;
}

// B 32x16 bf16 from Bt (N x K row-major): lane holds col n=lane&15,
// K range = k0 + (lane>>4)*16 .. +16 contiguous.
static __device__ __forceinline__ v16bf load_fragB(const __bf16* Bt, int ldb,
                                                   int n0, int k0) {
  const int lane = threadIdx.x & 31;
  const __bf16* p = Bt + (size_t)(n0 + (lane & 15)) * ldb + k0 + (lane >> 4) * 16;
  union { uint4 u[2]; v16bf f; } U;
  U.u[0] = reinterpret_cast<const uint4*>(p)[0];
  U.u[1] = reinterpret_cast<const uint4*>(p)[1];
  return U.f;
}

// Logits A: element (i,k) = qs[(r*NRES+i)*INNER + h*DH + d], r=k>>6, d=k&63.
// 8-element runs never cross an r-block (k0 multiple of 32).
static __device__ __forceinline__ v16bf load_fragA_dots(const __bf16* qs, int h,
                                                        int i0, int k0) {
  const int lane = threadIdx.x & 31;
  const int m = lane & 15, sel = lane >> 4;
  v16bf f;
#pragma unroll
  for (int g = 0; g < 2; ++g) {
    const int k = k0 + sel * 16 + g * 8;
    const int r = k >> 6, d = k & 63;
    const __bf16* p = qs + (size_t)(r * NRES + (i0 + m)) * INNER + h * DH + d;
    union { uint4 u; __bf16 h8[8]; } U;
    U.u = *reinterpret_cast<const uint4*>(p);
#pragma unroll
    for (int e = 0; e < 8; ++e) f[g * 8 + e] = U.h8[e];
  }
  return f;
}

// Logits B: element (j,k) = kv[(r*NRES+j)*(2*INNER) + h*DH + d]; 16 contiguous
// K per lane stays within one r-block (kbase multiple of 16).
static __device__ __forceinline__ v16bf load_fragB_dots(const __bf16* kv, int h,
                                                        int j0, int k0) {
  const int lane = threadIdx.x & 31;
  const int j = j0 + (lane & 15);
  const int k = k0 + (lane >> 4) * 16;
  const int r = k >> 6, d = k & 63;
  const __bf16* p = kv + (size_t)(r * NRES + j) * (2 * INNER) + h * DH + d;
  union { uint4 u[2]; v16bf f; } U;
  U.u[0] = reinterpret_cast<const uint4*>(p)[0];
  U.u[1] = reinterpret_cast<const uint4*>(p)[1];
  return U.f;
}

// attn@v B: element (n=d, k=j) = vbase[j*(2*INNER) + d]  (strided gather)
static __device__ __forceinline__ v16bf load_fragB_av(const __bf16* vbase,
                                                      int n0, int k0) {
  const int lane = threadIdx.x & 31;
  const int d = n0 + (lane & 15);
  const int j0 = k0 + (lane >> 4) * 16;
  v16bf f;
#pragma unroll
  for (int e = 0; e < 16; ++e)
    f[e] = vbase[(size_t)(j0 + e) * (2 * INNER) + d];
  return f;
}

// C/D 16x16 f32: col = n0 + (lane&15); rows m0 + (lane>>4)*8 + v
static __device__ __forceinline__ void store_tile_f32(float* C, int ldc, int m0,
                                                      int n0, v8f c,
                                                      const float* bias) {
  const int lane = threadIdx.x & 31;
  const int col = n0 + (lane & 15);
  const int row = m0 + (lane >> 4) * 8;
  const float bv = bias ? bias[col] : 0.0f;
#pragma unroll
  for (int v = 0; v < 8; ++v) C[(size_t)(row + v) * ldc + col] = c[v] + bv;
}

static __device__ __forceinline__ void store_tile_bf16(__bf16* C, int ldc, int m0,
                                                       int n0, v8f c,
                                                       const float* bias) {
  const int lane = threadIdx.x & 31;
  const int col = n0 + (lane & 15);
  const int row = m0 + (lane >> 4) * 8;
  const float bv = bias ? bias[col] : 0.0f;
#pragma unroll
  for (int v = 0; v < 8; ++v)
    C[(size_t)(row + v) * ldc + col] = (__bf16)(c[v] + bv);
}

// dispatch helper for the store (overload on pointer type) — declared BEFORE
// the template that calls it (no ADL for builtin pointer types).
static __device__ __forceinline__ void store_tile_f32_or_bf16(
    float* C, int ldc, int m0, int n0, v8f c, const float* bias) {
  store_tile_f32(C, ldc, m0, n0, c, bias);
}
static __device__ __forceinline__ void store_tile_f32_or_bf16(
    __bf16* C, int ldc, int m0, int n0, v8f c, const float* bias) {
  store_tile_bf16(C, ldc, m0, n0, c, bias);
}

// ---- Generic GEMM: C(MxN) = A(MxK) * Bt(NxK)^T ------------------------------
// (16*MT) x (16*NT) tile per wave. K must be a multiple of 64.
// Copy-free ping-pong pipeline: two fragment sets alternate, no register moves.
template <int MT, int NT, typename OutT>
__global__ void gemm_wmma_kernel(const __bf16* __restrict__ A, int lda,
                                 const __bf16* __restrict__ Bt, int ldb,
                                 OutT* __restrict__ C, int ldc, int M, int N,
                                 int K, const float* __restrict__ bias) {
  const int wave = blockIdx.x * (blockDim.x >> 5) + (threadIdx.x >> 5);
  const int ngrp = N / (16 * NT);
  const int m0 = (wave / ngrp) * (16 * MT);
  const int n0 = (wave % ngrp) * (16 * NT);
  if (m0 >= M) return;
  const int lane = threadIdx.x & 31;

  v8f acc[MT][NT] = {};
  v16bf a0[MT], b0[NT], a1[MT], b1[NT];
#pragma unroll
  for (int mi = 0; mi < MT; ++mi) a0[mi] = load_fragA(A, lda, m0 + mi * 16, 0);
#pragma unroll
  for (int t = 0; t < NT; ++t) b0[t] = load_fragB(Bt, ldb, n0 + t * 16, 0);

  for (int k0 = 0; k0 < K; k0 += 64) {
    // k0+32 always < K (K % 64 == 0): load set 1 unconditionally
#pragma unroll
    for (int mi = 0; mi < MT; ++mi)
      a1[mi] = load_fragA(A, lda, m0 + mi * 16, k0 + 32);
#pragma unroll
    for (int t = 0; t < NT; ++t)
      b1[t] = load_fragB(Bt, ldb, n0 + t * 16, k0 + 32);
#pragma unroll
    for (int mi = 0; mi < MT; ++mi)
#pragma unroll
      for (int t = 0; t < NT; ++t)
        acc[mi][t] = wmma_bf16(a0[mi], b0[t], acc[mi][t]);
    if (k0 + 64 < K) {
      __builtin_prefetch(A + (size_t)(m0 + (lane & 15)) * lda + k0 + 128, 0, 3);
#pragma unroll
      for (int mi = 0; mi < MT; ++mi)
        a0[mi] = load_fragA(A, lda, m0 + mi * 16, k0 + 64);
#pragma unroll
      for (int t = 0; t < NT; ++t)
        b0[t] = load_fragB(Bt, ldb, n0 + t * 16, k0 + 64);
    }
#pragma unroll
    for (int mi = 0; mi < MT; ++mi)
#pragma unroll
      for (int t = 0; t < NT; ++t)
        acc[mi][t] = wmma_bf16(a1[mi], b1[t], acc[mi][t]);
  }
#pragma unroll
  for (int mi = 0; mi < MT; ++mi)
#pragma unroll
    for (int t = 0; t < NT; ++t)
      store_tile_f32_or_bf16(C, ldc, m0 + mi * 16, n0 + t * 16, acc[mi][t],
                             bias);
}

// ---- Tied-row attention logits: dots[h][i][j] = sum_{r,d} qs * k ------------
// 32x32 tile per wave, ping-pong over K = RROWS*DH = 4096.
__global__ void attn_logits_kernel(const __bf16* __restrict__ qs,
                                   const __bf16* __restrict__ kv,
                                   float* __restrict__ dots) {
  const int h = blockIdx.z;
  const int wave = blockIdx.x * (blockDim.x >> 5) + (threadIdx.x >> 5); // 0..255
  const int i0 = (wave >> 4) * 32;
  const int j0 = (wave & 15) * 32;
  const int KTOT = RROWS * DH;

  v8f acc[2][2] = {};
  v16bf a0[2], b0[2], a1[2], b1[2];
#pragma unroll
  for (int mi = 0; mi < 2; ++mi)
    a0[mi] = load_fragA_dots(qs, h, i0 + mi * 16, 0);
#pragma unroll
  for (int t = 0; t < 2; ++t) b0[t] = load_fragB_dots(kv, h, j0 + t * 16, 0);

  for (int k0 = 0; k0 < KTOT; k0 += 64) {
#pragma unroll
    for (int mi = 0; mi < 2; ++mi)
      a1[mi] = load_fragA_dots(qs, h, i0 + mi * 16, k0 + 32);
#pragma unroll
    for (int t = 0; t < 2; ++t)
      b1[t] = load_fragB_dots(kv, h, j0 + t * 16, k0 + 32);
#pragma unroll
    for (int mi = 0; mi < 2; ++mi)
#pragma unroll
      for (int t = 0; t < 2; ++t)
        acc[mi][t] = wmma_bf16(a0[mi], b0[t], acc[mi][t]);
    if (k0 + 64 < KTOT) {
#pragma unroll
      for (int mi = 0; mi < 2; ++mi)
        a0[mi] = load_fragA_dots(qs, h, i0 + mi * 16, k0 + 64);
#pragma unroll
      for (int t = 0; t < 2; ++t)
        b0[t] = load_fragB_dots(kv, h, j0 + t * 16, k0 + 64);
    }
#pragma unroll
    for (int mi = 0; mi < 2; ++mi)
#pragma unroll
      for (int t = 0; t < 2; ++t)
        acc[mi][t] = wmma_bf16(a1[mi], b1[t], acc[mi][t]);
  }
  float* D = dots + (size_t)h * NRES * NRES;
#pragma unroll
  for (int mi = 0; mi < 2; ++mi)
#pragma unroll
    for (int t = 0; t < 2; ++t)
      store_tile_f32(D, NRES, i0 + mi * 16, j0 + t * 16, acc[mi][t], nullptr);
}

// ---- attn @ v, batched over (r,h); 32x32 tile per wave, ping-pong -----------
__global__ void attn_v_kernel(const __bf16* __restrict__ attn,
                              const __bf16* __restrict__ kv,
                              __bf16* __restrict__ ao) {
  const int z = blockIdx.z;  // r*8 + h
  const int r = z >> 3, h = z & 7;
  const int wave = blockIdx.x * (blockDim.x >> 5) + (threadIdx.x >> 5); // 0..31
  const int i0 = (wave >> 1) * 32;
  const int n0 = (wave & 1) * 32;
  const __bf16* A = attn + (size_t)h * NRES * NRES;
  const __bf16* vbase = kv + (size_t)r * NRES * (2 * INNER) + INNER + h * DH;

  v8f acc[2][2] = {};
  v16bf a0[2], b0[2], a1[2], b1[2];
#pragma unroll
  for (int mi = 0; mi < 2; ++mi) a0[mi] = load_fragA(A, NRES, i0 + mi * 16, 0);
#pragma unroll
  for (int t = 0; t < 2; ++t) b0[t] = load_fragB_av(vbase, n0 + t * 16, 0);

  for (int k0 = 0; k0 < NRES; k0 += 64) {
#pragma unroll
    for (int mi = 0; mi < 2; ++mi)
      a1[mi] = load_fragA(A, NRES, i0 + mi * 16, k0 + 32);
#pragma unroll
    for (int t = 0; t < 2; ++t)
      b1[t] = load_fragB_av(vbase, n0 + t * 16, k0 + 32);
#pragma unroll
    for (int mi = 0; mi < 2; ++mi)
#pragma unroll
      for (int t = 0; t < 2; ++t)
        acc[mi][t] = wmma_bf16(a0[mi], b0[t], acc[mi][t]);
    if (k0 + 64 < NRES) {
#pragma unroll
      for (int mi = 0; mi < 2; ++mi)
        a0[mi] = load_fragA(A, NRES, i0 + mi * 16, k0 + 64);
#pragma unroll
      for (int t = 0; t < 2; ++t)
        b0[t] = load_fragB_av(vbase, n0 + t * 16, k0 + 64);
    }
#pragma unroll
    for (int mi = 0; mi < 2; ++mi)
#pragma unroll
      for (int t = 0; t < 2; ++t)
        acc[mi][t] = wmma_bf16(a1[mi], b1[t], acc[mi][t]);
  }
  __bf16* O = ao + (size_t)r * NRES * INNER + h * DH;
#pragma unroll
  for (int mi = 0; mi < 2; ++mi)
#pragma unroll
    for (int t = 0; t < 2; ++t)
      store_tile_bf16(O, INNER, i0 + mi * 16, n0 + t * 16, acc[mi][t], nullptr);
}

// ---- Small elementwise / reduction kernels ---------------------------------
__global__ void f32_to_bf16_kernel(const float* __restrict__ src,
                                   __bf16* __restrict__ dst, int n) {
  for (int i = blockIdx.x * blockDim.x + threadIdx.x; i < n;
       i += gridDim.x * blockDim.x)
    dst[i] = (__bf16)src[i];
}

// dst (N x K) = transpose(src (K x N)), f32 -> bf16
__global__ void transpose_bf16_kernel(const float* __restrict__ src,
                                      __bf16* __restrict__ dst, int K, int N) {
  const int total = K * N;
  for (int i = blockIdx.x * blockDim.x + threadIdx.x; i < total;
       i += gridDim.x * blockDim.x) {
    const int nn = i / K, kk = i - nn * K;
    dst[i] = (__bf16)src[kk * N + nn];
  }
}

// w[i][h][r] = softmax_r( <sq[i,h*16:], sk[r,i,h*16:]> / 4 )
__global__ void row_weight_kernel(const float* __restrict__ sq,
                                  const __bf16* __restrict__ sk,
                                  float* __restrict__ w) {
  const int i = blockIdx.x >> 3, h = blockIdx.x & 7;
  const int r = threadIdx.x;  // 64 threads
  const float* sqp = sq + (size_t)i * DIMX + h * 16;
  const __bf16* skp = sk + (size_t)(r * NRES + i) * DIMX + h * 16;
  float dot = 0.f;
#pragma unroll
  for (int e = 0; e < 16; ++e) dot += sqp[e] * (float)skp[e];
  dot *= 0.25f;  // / sqrt(16)
  __shared__ float s[64];
  s[r] = dot;
  __syncthreads();
  for (int off = 32; off > 0; off >>= 1) {
    if (r < off) s[r] = fmaxf(s[r], s[r + off]);
    __syncthreads();
  }
  const float mx = s[0];
  __syncthreads();
  const float ex = __expf(dot - mx);
  s[r] = ex;
  __syncthreads();
  for (int off = 32; off > 0; off >>= 1) {
    if (r < off) s[r] += s[r + off];
    __syncthreads();
  }
  w[((size_t)i * HEADS + h) * RROWS + r] = ex / s[0];
}

// q[(r*N+i)*INNER + h*64+d] *= w[i][h][r] * dh^-0.5  (in place)
__global__ void scale_q_kernel(__bf16* __restrict__ q,
                               const float* __restrict__ w, int total) {
  for (int idx = blockIdx.x * blockDim.x + threadIdx.x; idx < total;
       idx += gridDim.x * blockDim.x) {
    const int row = idx >> 9;       // / INNER
    const int c = idx & 511;
    const int r = row >> 9;         // / NRES
    const int i = row & 511;
    const int h = c >> 6;
    const float v = (float)q[idx] * w[((size_t)i * HEADS + h) * RROWS + r] * 0.125f;
    q[idx] = (__bf16)v;
  }
}

// dots[h][i][j] += (LayerNorm(pair_bias[i][j]) @ Wpair)[h]; one wave per (i,j)
__global__ void pair_bias_kernel(const float* __restrict__ pb,
                                 const float* __restrict__ g,
                                 const float* __restrict__ beta,
                                 const float* __restrict__ Wp,
                                 float* __restrict__ dots) {
  const int wv = blockIdx.x * (blockDim.x >> 5) + (threadIdx.x >> 5);
  const int lane = threadIdx.x & 31;
  const int i = wv >> 9, j = wv & 511;
  const float* p = pb + ((size_t)i * NRES + j) * DPAIR + lane * 4;
  float x0 = p[0], x1 = p[1], x2 = p[2], x3 = p[3];
  float s = x0 + x1 + x2 + x3;
  for (int m = 16; m > 0; m >>= 1) s += __shfl_xor(s, m, 32);
  const float mu = s * (1.0f / 128.0f);
  const float d0 = x0 - mu, d1 = x1 - mu, d2 = x2 - mu, d3 = x3 - mu;
  float v = d0 * d0 + d1 * d1 + d2 * d2 + d3 * d3;
  for (int m = 16; m > 0; m >>= 1) v += __shfl_xor(v, m, 32);
  const float rstd = rsqrtf(v * (1.0f / 128.0f) + 1e-5f);
  const int c = lane * 4;
  const float n0 = d0 * rstd * g[c + 0] + beta[c + 0];
  const float n1 = d1 * rstd * g[c + 1] + beta[c + 1];
  const float n2 = d2 * rstd * g[c + 2] + beta[c + 2];
  const float n3 = d3 * rstd * g[c + 3] + beta[c + 3];
#pragma unroll
  for (int h = 0; h < HEADS; ++h) {
    float acc = n0 * Wp[(c + 0) * HEADS + h] + n1 * Wp[(c + 1) * HEADS + h] +
                n2 * Wp[(c + 2) * HEADS + h] + n3 * Wp[(c + 3) * HEADS + h];
    for (int m = 16; m > 0; m >>= 1) acc += __shfl_xor(acc, m, 32);
    if (lane == 0)
      dots[(size_t)h * NRES * NRES + (size_t)i * NRES + j] += acc;
  }
}

// softmax over j for one (h,i) row; 256 threads x 2 elems
__global__ void softmax_kernel(const float* __restrict__ dots,
                               __bf16* __restrict__ attn) {
  const int row = blockIdx.x;  // h*NRES + i
  const float* D = dots + (size_t)row * NRES;
  __bf16* A = attn + (size_t)row * NRES;
  const int t = threadIdx.x;
  const float a = D[t], b = D[t + 256];
  __shared__ float s[256];
  s[t] = fmaxf(a, b);
  __syncthreads();
  for (int off = 128; off > 0; off >>= 1) {
    if (t < off) s[t] = fmaxf(s[t], s[t + off]);
    __syncthreads();
  }
  const float mx = s[0];
  __syncthreads();
  const float ea = __expf(a - mx), eb = __expf(b - mx);
  s[t] = ea + eb;
  __syncthreads();
  for (int off = 128; off > 0; off >>= 1) {
    if (t < off) s[t] += s[t + off];
    __syncthreads();
  }
  const float inv = 1.0f / s[0];
  A[t] = (__bf16)(ea * inv);
  A[t + 256] = (__bf16)(eb * inv);
}

// ---- Launch -----------------------------------------------------------------
extern "C" void kernel_launch(void* const* d_in, const int* in_sizes, int n_in,
                              void* d_out, int out_size, void* d_ws,
                              size_t ws_size, hipStream_t stream) {
  (void)in_sizes; (void)n_in; (void)out_size; (void)ws_size;
  const float* x   = (const float*)d_in[0];
  const float* pb  = (const float*)d_in[1];
  const float* Wq  = (const float*)d_in[2];
  const float* Wkv = (const float*)d_in[3];
  const float* Wo  = (const float*)d_in[4];
  const float* bo  = (const float*)d_in[5];
  const float* lng = (const float*)d_in[6];
  const float* lnb = (const float*)d_in[7];
  const float* Wp  = (const float*)d_in[8];
  const float* Wsq = (const float*)d_in[9];
  const float* bsq = (const float*)d_in[10];
  const float* Wsk = (const float*)d_in[11];
  const float* bsk = (const float*)d_in[12];
  float* out = (float*)d_out;

  // Workspace carve (~158 MB total)
  char* ws = (char*)d_ws;
  size_t off = 0;
  auto carve = [&](size_t bytes) -> char* {
    char* p = ws + off;
    off += (bytes + 255) & ~(size_t)255;
    return p;
  };
  __bf16* xb    = (__bf16*)carve((size_t)MROWS * DIMX * 2);        // x bf16
  __bf16* WqT   = (__bf16*)carve((size_t)INNER * DIMX * 2);
  __bf16* WkvT  = (__bf16*)carve((size_t)2 * INNER * DIMX * 2);
  __bf16* WskT  = (__bf16*)carve((size_t)DIMX * DIMX * 2);
  __bf16* WsqT  = (__bf16*)carve((size_t)DIMX * DIMX * 2);
  __bf16* WoT   = (__bf16*)carve((size_t)DIMX * INNER * 2);
  float*  sqf   = (float*)carve((size_t)NRES * DIMX * 4);          // sq (r=0)
  float*  wbuf  = (float*)carve((size_t)NRES * HEADS * RROWS * 4); // row weights
  __bf16* skbuf = (__bf16*)carve((size_t)MROWS * DIMX * 2);
  __bf16* qbuf  = (__bf16*)carve((size_t)MROWS * INNER * 2);
  __bf16* kvbuf = (__bf16*)carve((size_t)MROWS * 2 * INNER * 2);
  float*  dots  = (float*)carve((size_t)HEADS * NRES * NRES * 4);
  __bf16* attn  = (__bf16*)carve((size_t)HEADS * NRES * NRES * 2);
  __bf16* aobuf = (__bf16*)carve((size_t)MROWS * INNER * 2);

  // 1) convert inputs to bf16 (weights transposed -> Bt layout)
  f32_to_bf16_kernel<<<1024, 256, 0, stream>>>(x, xb, MROWS * DIMX);
  transpose_bf16_kernel<<<256, 256, 0, stream>>>(Wq, WqT, DIMX, INNER);
  transpose_bf16_kernel<<<512, 256, 0, stream>>>(Wkv, WkvT, DIMX, 2 * INNER);
  transpose_bf16_kernel<<<64, 256, 0, stream>>>(Wsk, WskT, DIMX, DIMX);
  transpose_bf16_kernel<<<64, 256, 0, stream>>>(Wsq, WsqT, DIMX, DIMX);
  transpose_bf16_kernel<<<256, 256, 0, stream>>>(Wo, WoT, INNER, DIMX);

  // 2) projection GEMMs (WMMA, 32x32 tiles): q, kv, sk over all rows; sq r=0
  // waves = (M/32) * (N/32); 4 waves per 128-thread block
  gemm_wmma_kernel<2, 2, __bf16><<<4096, 128, 0, stream>>>(
      xb, DIMX, WqT, DIMX, qbuf, INNER, MROWS, INNER, DIMX, nullptr);
  gemm_wmma_kernel<2, 2, __bf16><<<8192, 128, 0, stream>>>(
      xb, DIMX, WkvT, DIMX, kvbuf, 2 * INNER, MROWS, 2 * INNER, DIMX, nullptr);
  gemm_wmma_kernel<2, 2, __bf16><<<1024, 128, 0, stream>>>(
      xb, DIMX, WskT, DIMX, skbuf, DIMX, MROWS, DIMX, DIMX, bsk);
  gemm_wmma_kernel<2, 2, float><<<16, 128, 0, stream>>>(
      xb, DIMX, WsqT, DIMX, sqf, DIMX, NRES, DIMX, DIMX, bsq);

  // 3) soft tied-row weights + fold (w * dh^-0.5) into q
  row_weight_kernel<<<NRES * HEADS, 64, 0, stream>>>(sqf, skbuf, wbuf);
  scale_q_kernel<<<4096, 256, 0, stream>>>(qbuf, wbuf, MROWS * INNER);

  // 4) tied logits (per-head 512x512 GEMM, K=4096) -> f32 dots
  // 256 waves per head / 4 per block = 64 blocks
  attn_logits_kernel<<<dim3(64, 1, HEADS), 128, 0, stream>>>(qbuf, kvbuf, dots);

  // 5) pair bias LN + projection, added into dots; then softmax over j
  pair_bias_kernel<<<32768, 256, 0, stream>>>(pb, lng, lnb, Wp, dots);
  softmax_kernel<<<HEADS * NRES, 256, 0, stream>>>(dots, attn);

  // 6) attn @ v batched over (r,h); 32 waves per (r,h) -> 8 blocks of 128
  attn_v_kernel<<<dim3(8, 1, RROWS * HEADS), 128, 0, stream>>>(attn, kvbuf,
                                                               aobuf);
  // final projection: (32768 x 512) @ WoT^T + bo -> f32 out
  gemm_wmma_kernel<2, 2, float><<<1024, 128, 0, stream>>>(
      aobuf, INNER, WoT, INNER, out, DIMX, MROWS, DIMX, INNER, bo);
}
